// CausalSelfAttentionGated_13657996001750
// MI455X (gfx1250) — matrix operand order repaired and checked by reading, not compiled, not run on hardware
//
#include <hip/hip_runtime.h>

// ---------------------------------------------------------------------------
// CausalSelfAttentionGated for MI455X (gfx1250, wave32, WMMA bf16)
// B=2 T=2048 C=2048 H=16 D=128 window=1024
// ---------------------------------------------------------------------------

typedef __attribute__((ext_vector_type(16))) __bf16 v16bf;
typedef __attribute__((ext_vector_type(8)))  float  v8f;

constexpr int Bv  = 2;
constexpr int Tt  = 2048;
constexpr int Cc  = 2048;
constexpr int Hh  = 16;
constexpr int Dd  = 128;
constexpr int WIN = 1024;

// Build a v16bf fragment from two aligned 16-byte chunks.
__device__ __forceinline__ v16bf load_frag2(const __bf16* p0, const __bf16* p1) {
    union { v16bf v; uint4 q[2]; } u;
    u.q[0] = *(const uint4*)p0;
    u.q[1] = *(const uint4*)p1;
    return u.v;
}

// Convert 8 floats -> 8 bf16 and store as one 16-byte chunk.
__device__ __forceinline__ void store8bf(__bf16* dst, const float* s) {
    union { __bf16 h[8]; uint4 q; } u;
#pragma unroll
    for (int j = 0; j < 8; ++j) u.h[j] = (__bf16)s[j];
    *(uint4*)dst = u.q;
}

// ---------------------------------------------------------------------------
// Tiled GEMM: C(MxN) = A(MxK) * W(KxN); W fp32, A fp32 or bf16.
// Block tile 128x128, BK=32, 256 threads = 8 waves (4M x 2N),
// each wave computes 32x64 via 2x4 WMMA 16x16x32 bf16 tiles.
// A staged row-major (row stride 40 -> A-frag = 2 x b128 loads);
// W staged transposed (WsT[n][k]  -> B-frag = 2 x b128 loads).
// OUT_MODE: 0 = fp32 row-major, 1 = bf16 row-major,
//           2 = bf16 scattered to (B,H,D,T)  (for V-transpose).
// ---------------------------------------------------------------------------
template <bool A_BF16, int OUT_MODE>
__global__ __launch_bounds__(256)
void gemm_wmma(const void* __restrict__ Ap, const float* __restrict__ W,
               void* __restrict__ Cp, int M, int N, int K)
{
    __shared__ __align__(16) __bf16 As[128][40];
    __shared__ __align__(16) __bf16 WsT[128][40];

    const int tid   = threadIdx.x;
    const int lane  = tid & 31;
    const int wid   = tid >> 5;
    const int waveM = wid >> 1;      // 0..3
    const int waveN = wid & 1;       // 0..1
    const int m0    = blockIdx.y * 128;
    const int n0    = blockIdx.x * 128;

    const float*  Af = (const float*)Ap;
    const __bf16* Ab = (const __bf16*)Ap;

    v8f acc[2][4] = {};

    for (int k0 = 0; k0 < K; k0 += 32) {
        // ---- stage A tile (128 x 32) ----
        {
            int row = tid >> 1;                 // 0..127
            int colb = (tid & 1) * 16;          // 0 or 16
#pragma unroll
            for (int g = 0; g < 2; ++g) {
                int col = colb + g * 8;
                size_t gidx = (size_t)(m0 + row) * K + k0 + col;
                if (A_BF16) {
                    *(uint4*)&As[row][col] = *(const uint4*)(Ab + gidx);
                } else {
                    float tmp[8];
                    float4 f0 = *(const float4*)(Af + gidx);
                    float4 f1 = *(const float4*)(Af + gidx + 4);
                    tmp[0] = f0.x; tmp[1] = f0.y; tmp[2] = f0.z; tmp[3] = f0.w;
                    tmp[4] = f1.x; tmp[5] = f1.y; tmp[6] = f1.z; tmp[7] = f1.w;
                    store8bf(&As[row][col], tmp);
                }
            }
        }
        // ---- stage W tile (32 x 128) transposed into WsT[n][k] ----
        {
            int row = tid >> 3;                 // k: 0..31
            int col = (tid & 7) * 16;           // n: 0,16,...,112
            size_t gidx = (size_t)(k0 + row) * N + n0 + col;
#pragma unroll
            for (int q = 0; q < 4; ++q) {
                float4 f = *(const float4*)(W + gidx + q * 4);
                WsT[col + q * 4 + 0][row] = (__bf16)f.x;
                WsT[col + q * 4 + 1][row] = (__bf16)f.y;
                WsT[col + q * 4 + 2][row] = (__bf16)f.z;
                WsT[col + q * 4 + 3][row] = (__bf16)f.w;
            }
        }
        if (k0 + 32 < K)
            __builtin_prefetch(W + (size_t)(k0 + 32) * N + n0, 0, 1);
        __syncthreads();

        // ---- fragments + WMMA ----
        v16bf af[2], bfr[4];
#pragma unroll
        for (int mt = 0; mt < 2; ++mt) {
            int row = waveM * 32 + mt * 16 + (lane & 15);
            int kb  = (lane >> 4) << 4;         // 0 or 16
            af[mt] = load_frag2(&As[row][kb], &As[row][kb + 8]);
        }
#pragma unroll
        for (int nt = 0; nt < 4; ++nt) {
            int col = waveN * 64 + nt * 16 + (lane & 15);
            int kb  = (lane >> 4) << 4;         // 0 or 16
            bfr[nt] = load_frag2(&WsT[col][kb], &WsT[col][kb + 8]);
        }
#pragma unroll
        for (int mt = 0; mt < 2; ++mt)
#pragma unroll
            for (int nt = 0; nt < 4; ++nt)
                acc[mt][nt] = __builtin_amdgcn_wmma_f32_16x16x32_bf16(
                    false, af[mt], false, bfr[nt], (short)0, acc[mt][nt],
                    false, false);
        __syncthreads();
    }

    // ---- epilogue (C layout: VGPR r -> M=r (+8 lanes 16-31), N=lane&15) ----
#pragma unroll
    for (int mt = 0; mt < 2; ++mt)
#pragma unroll
        for (int nt = 0; nt < 4; ++nt)
#pragma unroll
            for (int r = 0; r < 8; ++r) {
                int row = m0 + waveM * 32 + mt * 16 + r + ((lane >> 4) << 3);
                int col = n0 + waveN * 64 + nt * 16 + (lane & 15);
                float v = acc[mt][nt][r];
                if (OUT_MODE == 0) {
                    ((float*)Cp)[(size_t)row * N + col] = v;
                } else if (OUT_MODE == 1) {
                    ((__bf16*)Cp)[(size_t)row * N + col] = (__bf16)v;
                } else {
                    // scatter to (B,H,D,T): row = b*T+t, col = h*128+d
                    int bb = row >> 11, t = row & (Tt - 1);
                    int hh = col >> 7,  d = col & (Dd - 1);
                    ((__bf16*)Cp)[(((size_t)bb * Hh + hh) * Dd + d) * Tt + t] =
                        (__bf16)v;
                }
            }
}

// ---------------------------------------------------------------------------
// In-place RoPE + RMS-norm on a (B*T*H, 128) bf16 buffer.
// ---------------------------------------------------------------------------
__global__ __launch_bounds__(128)
void rope_rms_kernel(__bf16* __restrict__ buf, const float* __restrict__ cosp,
                     const float* __restrict__ sinp)
{
    int gw   = blockIdx.x * 4 + (threadIdx.x >> 5);   // (b*T + t)*H + h
    int lane = threadIdx.x & 31;
    int t    = (gw >> 4) & (Tt - 1);

    __bf16* rp = buf + (size_t)gw * Dd;
    float x0 = (float)rp[lane];
    float x1 = (float)rp[lane + 32];
    float x2 = (float)rp[lane + 64];
    float x3 = (float)rp[lane + 96];

    float c0 = cosp[t * 64 + lane],      s0 = sinp[t * 64 + lane];
    float c1 = cosp[t * 64 + lane + 32], s1 = sinp[t * 64 + lane + 32];

    float o0 =  x0 * c0 + x2 * s0;
    float o2 = -x0 * s0 + x2 * c0;
    float o1 =  x1 * c1 + x3 * s1;
    float o3 = -x1 * s1 + x3 * c1;

    float ss = o0 * o0 + o1 * o1 + o2 * o2 + o3 * o3;
#pragma unroll
    for (int msk = 1; msk < 32; msk <<= 1)
        ss += __shfl_xor(ss, msk, 32);
    float rn = rsqrtf(ss * (1.0f / 128.0f) + 1.1920929e-07f);

    rp[lane]      = (__bf16)(o0 * rn);
    rp[lane + 32] = (__bf16)(o1 * rn);
    rp[lane + 64] = (__bf16)(o2 * rn);
    rp[lane + 96] = (__bf16)(o3 * rn);
}

// ---------------------------------------------------------------------------
// Forget-gate: lam = elu(x@wl)+1 ; z = (x@fw + fb)*lam ;
// logf[b,h,t] = log_sigmoid(z) / (lam + 0.001).
// ---------------------------------------------------------------------------
__global__ __launch_bounds__(32)
void gates_kernel(const float* __restrict__ x, const float* __restrict__ fw,
                  const float* __restrict__ fb, const float* __restrict__ wl,
                  float* __restrict__ logf)
{
    int row  = blockIdx.x;          // b*T + t
    int lane = threadIdx.x;
    int h    = lane & 15;
    const float* Wsel = (lane < 16) ? fw : wl;
    const float* xr   = x + (size_t)row * Cc;

    float acc = 0.f;
    for (int k = 0; k < Cc; k += 4) {
        float4 xv = *(const float4*)(xr + k);
        acc += xv.x * Wsel[(k + 0) * Hh + h];
        acc += xv.y * Wsel[(k + 1) * Hh + h];
        acc += xv.z * Wsel[(k + 2) * Hh + h];
        acc += xv.w * Wsel[(k + 3) * Hh + h];
    }
    float other = __shfl_down(acc, 16, 32);
    if (lane < 16) {
        float lam = (other > 0.f) ? (other + 1.f) : __expf(other);
        float z   = (acc + fb[h]) * lam;
        float ls  = fminf(z, 0.f) - log1pf(__expf(-fabsf(z)));
        int b = row >> 11, t = row & (Tt - 1);
        logf[((size_t)(b * Hh + h)) * Tt + t] = ls / (lam + 0.001f);
    }
}

// ---------------------------------------------------------------------------
// Inclusive cumsum over T per (b,h).
// ---------------------------------------------------------------------------
__global__ __launch_bounds__(256)
void scan_kernel(const float* __restrict__ logf, float* __restrict__ cumF)
{
    __shared__ float sums[256];
    int tid = threadIdx.x;
    size_t base = (size_t)blockIdx.x * Tt;

    float v[8], pref[8];
    float run = 0.f;
#pragma unroll
    for (int j = 0; j < 8; ++j) {
        v[j] = logf[base + tid * 8 + j];
        run += v[j];
        pref[j] = run;
    }
    sums[tid] = run;
    __syncthreads();
    for (int off = 1; off < 256; off <<= 1) {
        float add = (tid >= off) ? sums[tid - off] : 0.f;
        __syncthreads();
        sums[tid] += add;
        __syncthreads();
    }
    float excl = sums[tid] - run;
#pragma unroll
    for (int j = 0; j < 8; ++j)
        cumF[base + tid * 8 + j] = excl + pref[j];
}

// ---------------------------------------------------------------------------
// Windowed-causal flash attention with gate bias.
// One wave owns 16 query rows; k-tiles of 32; D=128.
// q,k bf16 (B,T,H,D); v bf16 pre-transposed to (B,H,D,T) so the P*V
// B-fragment is a single aligned 32-byte global load.
// P round-trips per-wave LDS (row stride 40 -> 2 x b128 reads).
// ---------------------------------------------------------------------------
__global__ __launch_bounds__(128)
void attn_kernel(const __bf16* __restrict__ qb, const __bf16* __restrict__ kb,
                 const __bf16* __restrict__ vt, const float* __restrict__ cumF,
                 __bf16* __restrict__ yb)
{
    __shared__ __align__(16) __bf16 Pst[4][16][40];

    const int tid  = threadIdx.x;
    const int lane = tid & 31;
    const int wid  = tid >> 5;
    const int h    = blockIdx.y;
    const int b    = blockIdx.z;
    const int q0   = (blockIdx.x * 4 + wid) * 16;

    const float SCALE = 0.08838834764831845f;   // 1/sqrt(128)
    const float NEGM  = -1e38f;

    // Q fragments (A-layout), 4 chunks of K=32 over D=128
    v16bf qa[4];
    {
        const __bf16* qp =
            qb + ((size_t)(b * Tt + q0 + (lane & 15)) * Hh + h) * Dd +
            ((lane >> 4) << 4);
#pragma unroll
        for (int c = 0; c < 4; ++c)
            qa[c] = load_frag2(qp + c * 32, qp + c * 32 + 8);
    }

    const float* cf = cumF + (size_t)(b * Hh + h) * Tt;
    float cq[8], m[8], l[8];
#pragma unroll
    for (int r = 0; r < 8; ++r) {
        int i = q0 + r + ((lane >> 4) << 3);
        cq[r] = cf[i];
        m[r]  = NEGM;
        l[r]  = 0.f;
    }
    v8f acc[8] = {};

    int klo = q0 - (WIN - 1);
    if (klo < 0) klo = 0;
    klo &= ~31;

    const __bf16* vbase =
        vt + ((size_t)(b * Hh + h) * Dd + (lane & 15)) * Tt + ((lane >> 4) << 4);

    for (int kt = klo; kt <= q0 + 15; kt += 32) {
        // ---- S = Q * K^T for 32 keys (two 16x16 tiles) ----
        v8f S0 = {}, S1 = {};
        {
            int kr0 = kt + (lane & 15);       if (kr0 > Tt - 1) kr0 = Tt - 1;
            int kr1 = kt + 16 + (lane & 15);  if (kr1 > Tt - 1) kr1 = Tt - 1;
            const __bf16* kp0 =
                kb + ((size_t)(b * Tt + kr0) * Hh + h) * Dd + ((lane >> 4) << 4);
            const __bf16* kp1 =
                kb + ((size_t)(b * Tt + kr1) * Hh + h) * Dd + ((lane >> 4) << 4);
#pragma unroll
            for (int c = 0; c < 4; ++c) {
                v16bf k0v = *(const v16bf*)(kp0 + c * 32);
                S0 = __builtin_amdgcn_wmma_f32_16x16x32_bf16(
                    false, qa[c], false, k0v, (short)0, S0, false, false);
                v16bf k1v = *(const v16bf*)(kp1 + c * 32);
                S1 = __builtin_amdgcn_wmma_f32_16x16x32_bf16(
                    false, qa[c], false, k1v, (short)0, S1, false, false);
            }
        }

        // ---- bias + mask + online softmax ----
        int j0 = kt + (lane & 15), j1 = j0 + 16;
        float ck0 = cf[j0 < Tt ? j0 : Tt - 1];
        float ck1 = cf[j1 < Tt ? j1 : Tt - 1];
        float alph[8];
#pragma unroll
        for (int r = 0; r < 8; ++r) {
            int i = q0 + r + ((lane >> 4) << 3);
            float s0 = S0[r] * SCALE + cq[r] - ck0;
            float s1 = S1[r] * SCALE + cq[r] - ck1;
            s0 = ((j0 <= i) && (i - j0 < WIN)) ? s0 : NEGM;
            s1 = ((j1 <= i) && (i - j1 < WIN)) ? s1 : NEGM;
            float rm = fmaxf(s0, s1);
#pragma unroll
            for (int msk = 1; msk < 16; msk <<= 1)
                rm = fmaxf(rm, __shfl_xor(rm, msk, 32));
            float newm  = fmaxf(m[r], rm);
            float msafe = fmaxf(newm, -1e30f);
            float a     = __expf(m[r] - msafe);
            m[r] = newm;
            float p0 = __expf(s0 - msafe);
            float p1 = __expf(s1 - msafe);
            int Mr = r + ((lane >> 4) << 3);
            Pst[wid][Mr][lane & 15]        = (__bf16)p0;
            Pst[wid][Mr][16 + (lane & 15)] = (__bf16)p1;
            float rs = p0 + p1;
#pragma unroll
            for (int msk = 1; msk < 16; msk <<= 1)
                rs += __shfl_xor(rs, msk, 32);
            l[r]    = l[r] * a + rs;
            alph[r] = a;
        }
#pragma unroll
        for (int dt = 0; dt < 8; ++dt)
#pragma unroll
            for (int r = 0; r < 8; ++r)
                acc[dt][r] *= alph[r];

        // ---- y += P * V  (V fragments straight from transposed global) ----
        v16bf pa;
        {
            int kb2 = (lane >> 4) << 4;
            pa = load_frag2(&Pst[wid][lane & 15][kb2],
                            &Pst[wid][lane & 15][kb2 + 8]);
        }
#pragma unroll
        for (int dt = 0; dt < 8; ++dt) {
            v16bf vf = *(const v16bf*)(vbase + (size_t)(dt * 16) * Tt + kt);
            acc[dt] = __builtin_amdgcn_wmma_f32_16x16x32_bf16(
                false, pa, false, vf, (short)0, acc[dt], false, false);
        }
    }

    // ---- normalize + store y (bf16, (B,T,H,D)) ----
    float linv[8];
#pragma unroll
    for (int r = 0; r < 8; ++r) linv[r] = 1.f / l[r];
#pragma unroll
    for (int dt = 0; dt < 8; ++dt)
#pragma unroll
        for (int r = 0; r < 8; ++r) {
            int i = q0 + r + ((lane >> 4) << 3);
            int d = dt * 16 + (lane & 15);
            yb[((size_t)(b * Tt + i) * Hh + h) * Dd + d] =
                (__bf16)(acc[dt][r] * linv[r]);
        }
}

// ---------------------------------------------------------------------------
// Host launcher
// ---------------------------------------------------------------------------
extern "C" void kernel_launch(void* const* d_in, const int* in_sizes, int n_in,
                              void* d_out, int out_size, void* d_ws,
                              size_t ws_size, hipStream_t stream)
{
    const float* x    = (const float*)d_in[0];
    const float* cosp = (const float*)d_in[1];
    const float* sinp = (const float*)d_in[2];
    const float* Wq   = (const float*)d_in[3];
    const float* Wk   = (const float*)d_in[4];
    const float* Wv   = (const float*)d_in[5];
    const float* Wo   = (const float*)d_in[6];
    const float* fw   = (const float*)d_in[7];
    const float* fb   = (const float*)d_in[8];
    const float* wl   = (const float*)d_in[9];
    float* out = (float*)d_out;

    const size_t BT   = (size_t)Bv * Tt;        // 4096
    const size_t ROWS = BT * (size_t)Cc;        // 8M elements

    char* p = (char*)d_ws;
    auto alloc = [&](size_t bytes) -> char* {
        char* r = p;
        p += (bytes + 255) & ~(size_t)255;
        return r;
    };
    __bf16* qbf  = (__bf16*)alloc(ROWS * 2);
    __bf16* kbf  = (__bf16*)alloc(ROWS * 2);
    __bf16* vtb  = (__bf16*)alloc(ROWS * 2);    // (B,H,D,T)
    __bf16* ybf  = (__bf16*)alloc(ROWS * 2);
    float*  logf = (float*)alloc((size_t)Bv * Hh * Tt * 4);
    float*  cumF = (float*)alloc((size_t)Bv * Hh * Tt * 4);

    dim3 gblk(Cc / 128, BT / 128);   // (16, 32)

    // QKV projections (bf16 outputs; V scattered to (B,H,D,T))
    gemm_wmma<false, 1><<<gblk, 256, 0, stream>>>(x, Wq, qbf, (int)BT, Cc, Cc);
    gemm_wmma<false, 1><<<gblk, 256, 0, stream>>>(x, Wk, kbf, (int)BT, Cc, Cc);
    gemm_wmma<false, 2><<<gblk, 256, 0, stream>>>(x, Wv, vtb, (int)BT, Cc, Cc);

    // forget-gate logits + cumulative sum
    gates_kernel<<<(int)BT, 32, 0, stream>>>(x, fw, fb, wl, logf);
    scan_kernel<<<Bv * Hh, 256, 0, stream>>>(logf, cumF);

    // RoPE + RMS-norm (in place) on q and k
    rope_rms_kernel<<<(int)((BT * Hh) / 4), 128, 0, stream>>>(qbf, cosp, sinp);
    rope_rms_kernel<<<(int)((BT * Hh) / 4), 128, 0, stream>>>(kbf, cosp, sinp);

    // windowed causal gated attention
    attn_kernel<<<dim3(Tt / 64, Hh, Bv), 128, 0, stream>>>(qbf, kbf, vtb, cumF,
                                                           ybf);

    // output projection (fp32 out)
    gemm_wmma<true, 0><<<gblk, 256, 0, stream>>>(ybf, Wo, out, (int)BT, Cc, Cc);
}
